// Head_28527172780351
// MI455X (gfx1250) — matrix operand order repaired and checked by reading, not compiled, not run on hardware
//
#include <hip/hip_runtime.h>
#include <hip/hip_fp16.h>

typedef __attribute__((ext_vector_type(16))) _Float16 v16h;
typedef __attribute__((ext_vector_type(8)))  _Float16 v8h;
typedef __attribute__((ext_vector_type(4)))  _Float16 v4h;
typedef __attribute__((ext_vector_type(8)))  float    v8f;
typedef __attribute__((__vector_size__(4 * sizeof(int)))) int v4i;

#define B_ 16
#define T_ 2048
#define C_ 1024
#define D_ 128
#define SCALE 0.08838834764831845f  // 128^-0.5

// Build a 16-half WMMA fragment register from two contiguous 8-half runs.
static __device__ __forceinline__ v16h ld16(const _Float16* lo, const _Float16* hi) {
    v8h a = *(const v8h*)lo;
    v8h b = *(const v8h*)hi;
    v16h r;
#pragma unroll
    for (int i = 0; i < 8; ++i) { r[i] = a[i]; r[i + 8] = b[i]; }
    return r;
}

static __device__ __forceinline__ float rmax16(float v) {
#pragma unroll
    for (int off = 1; off < 16; off <<= 1) v = fmaxf(v, __shfl_xor(v, off, 16));
    return v;
}
static __device__ __forceinline__ float rsum16(float v) {
#pragma unroll
    for (int off = 1; off < 16; off <<= 1) v += __shfl_xor(v, off, 16);
    return v;
}

// ---- CDNA5 async global->LDS copy (ASYNCcnt path), with safe fallback -----
static __device__ __forceinline__ void async_cp16B(const _Float16* g, _Float16* l) {
#if __has_builtin(__builtin_amdgcn_global_load_async_to_lds_b128)
    __builtin_amdgcn_global_load_async_to_lds_b128(
        (__attribute__((address_space(1))) v4i*)(void*)g,
        (__attribute__((address_space(3))) v4i*)(void*)l, 0, 0);
#else
    *(v8h*)l = *(const v8h*)g;
#endif
}
static __device__ __forceinline__ void wait_async0() {
#if __has_builtin(__builtin_amdgcn_s_wait_asynccnt)
    __builtin_amdgcn_s_wait_asynccnt(0);
#else
    asm volatile("s_wait_asynccnt 0" ::: "memory");
#endif
}

// ---------------------------------------------------------------------------
// Kernel 1: one projection  out_f16[M=B*T, D] = f16( x[M, C] @ W[C, D] )
// Block: 256 threads = 8 waves. Block tile: 128 rows x 128 cols.
// Wave tile: 16 rows x 128 cols (8 WMMA accumulators), K-loop step 32.
// ---------------------------------------------------------------------------
__global__ __launch_bounds__(256) void qkv_proj_kernel(
    const float* __restrict__ x, const float* __restrict__ W,
    _Float16* __restrict__ out)
{
    __shared__ __align__(16) _Float16 lx[128 * 32];   // x tile  [row][k]
    __shared__ __align__(16) _Float16 lw[128 * 32];   // W tile  [n][k] (transposed)

    const int tid  = threadIdx.x;
    const int lane = tid & 31;
    const int wave = tid >> 5;          // 0..7
    const int g    = lane >> 4;         // half-wave selector
    const int ln   = lane & 15;
    const long mbase = (long)blockIdx.x * 128;

    v8f acc[8];
#pragma unroll
    for (int nt = 0; nt < 8; ++nt)
#pragma unroll
        for (int r = 0; r < 8; ++r) acc[nt][r] = 0.0f;

    for (int ks = 0; ks < C_; ks += 32) {
        // stage x tile: 128x32 f32 -> f16 LDS, float4 loads / v4h stores
#pragma unroll
        for (int i = 0; i < 4; ++i) {
            int c4 = tid + i * 256;              // 0..1023 float4 chunks
            int r  = c4 >> 3;
            int kq = (c4 & 7) * 4;
            float4 f = *(const float4*)&x[(mbase + r) * C_ + ks + kq];
            v4h h; h[0] = (_Float16)f.x; h[1] = (_Float16)f.y;
                   h[2] = (_Float16)f.z; h[3] = (_Float16)f.w;
            *(v4h*)&lx[r * 32 + kq] = h;
        }
        // stage W tile transposed: W[ks+k][n] -> lw[n][k], float4 loads
#pragma unroll
        for (int i = 0; i < 4; ++i) {
            int c4 = tid + i * 256;              // 0..1023
            int k  = c4 >> 5;                    // 0..31
            int n4 = (c4 & 31) * 4;              // 0..124
            float4 f = *(const float4*)&W[(long)(ks + k) * D_ + n4];
            lw[(n4 + 0) * 32 + k] = (_Float16)f.x;
            lw[(n4 + 1) * 32 + k] = (_Float16)f.y;
            lw[(n4 + 2) * 32 + k] = (_Float16)f.z;
            lw[(n4 + 3) * 32 + k] = (_Float16)f.w;
        }
        if (ks + 32 < C_)   // speculative prefetch of the next x tile
            __builtin_prefetch(&x[(mbase + (tid >> 1)) * C_ + ks + 32], 0, 1);
        __syncthreads();

        // A fragment: 16x32 f16, row = wave*16 + ln
        const _Float16* arow = &lx[(wave * 16 + ln) * 32];
        v16h a = ld16(arow + g * 16, arow + g * 16 + 8);

        // batch-load all 8 B fragments, then issue the WMMAs back-to-back
        v16h bfr[8];
#pragma unroll
        for (int nt = 0; nt < 8; ++nt) {
            const _Float16* bcol = &lw[(nt * 16 + ln) * 32 + g * 16];
            bfr[nt] = ld16(bcol, bcol + 8);
        }
#pragma unroll
        for (int nt = 0; nt < 8; ++nt)
            acc[nt] = __builtin_amdgcn_wmma_f32_16x16x32_f16(
                false, a, false, bfr[nt], (short)0, acc[nt], false, false);
        __syncthreads();
    }

    // C layout: vgpr r, lanes 0-15 -> row r; lanes 16-31 -> row 8+r; col = ln
    const long row0 = mbase + wave * 16 + g * 8;
#pragma unroll
    for (int nt = 0; nt < 8; ++nt)
#pragma unroll
        for (int r = 0; r < 8; ++r)
            out[(row0 + r) * D_ + nt * 16 + ln] = (_Float16)acc[nt][r];
}

// ---------------------------------------------------------------------------
// Kernel 2: flash attention, block-causal.
// Grid: (T/64, B). Block: 128 threads = 4 waves; wave w owns 16 query rows.
// ---------------------------------------------------------------------------
__global__ __launch_bounds__(128) void attn_kernel(
    const _Float16* __restrict__ Qf, const _Float16* __restrict__ Kf,
    const _Float16* __restrict__ Vf, float* __restrict__ out)
{
    __shared__ __align__(16) _Float16 lk [64 * 128];   // K block [key][d]
    __shared__ __align__(16) _Float16 lv [64 * 128];   // V block [key][d] (async landing)
    __shared__ __align__(16) _Float16 lvT[128 * 64];   // V block [d][key]
    __shared__ __align__(16) _Float16 lp [4][16 * 64]; // per-wave P [qrow][key]

    const int tid  = threadIdx.x;
    const int lane = tid & 31;
    const int w    = tid >> 5;             // wave 0..3
    const int g    = lane >> 4;
    const int ln   = lane & 15;
    const int qb   = blockIdx.x;
    const int b    = blockIdx.y;
    const long base = (long)b * T_;        // row base into [B*T, D] arrays
    const int qrow0 = qb * 64 + w * 16;

    // Persistent Q fragments: 16 rows x 128 d = 4 chunks of 16x32
    v16h qfrag[4];
    {
        const _Float16* qrow = Qf + (base + qrow0 + ln) * D_;
#pragma unroll
        for (int kc = 0; kc < 4; ++kc)
            qfrag[kc] = ld16(qrow + kc * 32 + g * 16, qrow + kc * 32 + g * 16 + 8);
    }

    v8f o[8];
    float m[8], l[8];
#pragma unroll
    for (int nt = 0; nt < 8; ++nt)
#pragma unroll
        for (int r = 0; r < 8; ++r) o[nt][r] = 0.0f;
#pragma unroll
    for (int r = 0; r < 8; ++r) { m[r] = -1.0e30f; l[r] = 0.0f; }

    for (int kb = 0; kb <= qb; ++kb) {
        const long kr0 = base + kb * 64;

        __syncthreads();   // previous iteration's consumers are done

        // async-stage K and V blocks (1024 x 16B chunks each, 8 per thread)
#pragma unroll
        for (int i = 0; i < 8; ++i) {
            int chunk = tid + i * 128;          // 0..1023
            int r  = chunk >> 4;
            int dc = (chunk & 15) * 8;
            async_cp16B(&Kf[(kr0 + r) * D_ + dc], &lk[r * 128 + dc]);
            async_cp16B(&Vf[(kr0 + r) * D_ + dc], &lv[r * 128 + dc]);
        }
        if (kb + 1 <= qb)
            __builtin_prefetch(&Kf[(kr0 + 64 + (tid >> 1)) * D_], 0, 1);
        wait_async0();
        __syncthreads();   // lk / lv visible to all waves

        // LDS->LDS transpose of V for the PV B-fragments
#pragma unroll
        for (int i = 0; i < 8; ++i) {
            int chunk = tid + i * 128;
            int r  = chunk >> 4;
            int dc = (chunk & 15) * 8;
            v8h vv = *(const v8h*)&lv[r * 128 + dc];
#pragma unroll
            for (int j = 0; j < 8; ++j) lvT[(dc + j) * 64 + r] = vv[j];
        }

        const bool diag = (kb == qb);
        if (diag) {
            // tiles kt > w are fully masked and skipped: zero their P region
            for (int kt = w + 1; kt < 4; ++kt)
#pragma unroll
                for (int r = 0; r < 8; ++r)
                    lp[w][(g * 8 + r) * 64 + kt * 16 + ln] = (_Float16)0.0f;
        }

        for (int kt = 0; kt < 4; ++kt) {
            if (diag && kt > w) break;

            // S = Q K^T for 16 queries x 16 keys: batch fragment loads, then WMMA
            v16h bk[4];
#pragma unroll
            for (int kc = 0; kc < 4; ++kc) {
                const _Float16* p = &lk[(kt * 16 + ln) * 128 + kc * 32 + g * 16];
                bk[kc] = ld16(p, p + 8);
            }
            v8f s;
#pragma unroll
            for (int r = 0; r < 8; ++r) s[r] = 0.0f;
#pragma unroll
            for (int kc = 0; kc < 4; ++kc)
                s = __builtin_amdgcn_wmma_f32_16x16x32_f16(
                    false, qfrag[kc], false, bk[kc], (short)0, s, false, false);
#pragma unroll
            for (int r = 0; r < 8; ++r) s[r] *= SCALE;

            if (diag) {
                int key = kb * 64 + kt * 16 + ln;
#pragma unroll
                for (int r = 0; r < 8; ++r) {
                    int qr = qrow0 + g * 8 + r;
                    if (key > qr) s[r] = -1.0e30f;
                }
            }

            // online softmax update (rows live across the 16-lane half-wave)
            float alpha[8];
#pragma unroll
            for (int r = 0; r < 8; ++r) {
                float mx = rmax16(s[r]);
                float mn = fmaxf(m[r], mx);
                alpha[r] = __expf(m[r] - mn);
                float p  = __expf(s[r] - mn);
                l[r] = l[r] * alpha[r] + rsum16(p);
                m[r] = mn;
                lp[w][(g * 8 + r) * 64 + kt * 16 + ln] = (_Float16)p;
            }
#pragma unroll
            for (int nt = 0; nt < 8; ++nt)
#pragma unroll
                for (int r = 0; r < 8; ++r) o[nt][r] *= alpha[r];
        }

        __syncthreads();   // all transposes + P writes complete before PV

        // O += P V : A = P (16 x 64 keys, two 16x32 chunks), B = V^T from lvT
#pragma unroll
        for (int kc2 = 0; kc2 < 2; ++kc2) {
            const _Float16* prow = &lp[w][ln * 64 + kc2 * 32 + g * 16];
            v16h ap = ld16(prow, prow + 8);
            v16h bv[8];
#pragma unroll
            for (int nt = 0; nt < 8; ++nt) {
                const _Float16* vcol = &lvT[(nt * 16 + ln) * 64 + kc2 * 32 + g * 16];
                bv[nt] = ld16(vcol, vcol + 8);
            }
#pragma unroll
            for (int nt = 0; nt < 8; ++nt)
                o[nt] = __builtin_amdgcn_wmma_f32_16x16x32_f16(
                    false, ap, false, bv[nt], (short)0, o[nt], false, false);
        }
    }

    // normalize and store fp32 output
    float li[8];
#pragma unroll
    for (int r = 0; r < 8; ++r) li[r] = 1.0f / l[r];
#pragma unroll
    for (int nt = 0; nt < 8; ++nt)
#pragma unroll
        for (int r = 0; r < 8; ++r)
            out[(base + qrow0 + g * 8 + r) * D_ + nt * 16 + ln] = o[nt][r] * li[r];
}

// ---------------------------------------------------------------------------
extern "C" void kernel_launch(void* const* d_in, const int* in_sizes, int n_in,
                              void* d_out, int out_size, void* d_ws, size_t ws_size,
                              hipStream_t stream) {
    const float* x  = (const float*)d_in[0];
    const float* Wq = (const float*)d_in[1];
    const float* Wk = (const float*)d_in[2];
    const float* Wv = (const float*)d_in[3];
    float* out = (float*)d_out;

    // f16 Q/K/V intermediates in workspace: 3 * B*T*D * 2 bytes = 24 MiB
    _Float16* Qf = (_Float16*)d_ws;
    _Float16* Kf = Qf + (size_t)B_ * T_ * D_;
    _Float16* Vf = Kf + (size_t)B_ * T_ * D_;

    const dim3 gp((B_ * T_) / 128);
    qkv_proj_kernel<<<gp, 256, 0, stream>>>(x, Wq, Qf);
    qkv_proj_kernel<<<gp, 256, 0, stream>>>(x, Wk, Kf);
    qkv_proj_kernel<<<gp, 256, 0, stream>>>(x, Wv, Vf);

    attn_kernel<<<dim3(T_ / 64, B_), 128, 0, stream>>>(Qf, Kf, Vf, out);
}
